// CRF_62019327754573
// MI455X (gfx1250) — matrix operand order repaired and checked by reading, not compiled, not run on hardware
//
#include <hip/hip_runtime.h>

#define TLEN 1024
#define BATCH 256
#define K0 50
#define START_TAG 48
#define STOP_TAG 49

typedef __attribute__((ext_vector_type(16))) __bf16 v16bf;
typedef __attribute__((ext_vector_type(8)))  float  v8f;

union BF16x16 { unsigned int u[8]; v16bf v; };

// round-to-nearest-even f32 -> bf16 bits (init-time only; no-NaN inputs)
__device__ __forceinline__ unsigned int f2bf_bits(float x) {
  unsigned int u = __float_as_uint(x);
  return (u + 0x7FFFu + ((u >> 16) & 1u)) >> 16;
}
__device__ __forceinline__ float bfbits2f(unsigned int b) {
  return __uint_as_float(b << 16);
}

// hot-path pack: plain __bf16 casts -> compiler can pick v_cvt_pk_bf16_f32
__device__ __forceinline__ unsigned int pack_bf16(float x, float y) {
  union { __bf16 b[2]; unsigned int u; } c;
  c.b[0] = (__bf16)x;
  c.b[1] = (__bf16)y;
  return c.u;
}

__global__ void __launch_bounds__(32)
crf_forward_kernel(const float* __restrict__ feats,
                   const float* __restrict__ trans,
                   const unsigned char* __restrict__ mask,
                   float* __restrict__ partials)
{
  // alpha^T staging tile: [b (16), i (64)] column-major, stride 66 floats
  // -> stores hit distinct banks ((2b+i)%64), b64 loads stay 8B-aligned.
  __shared__ __align__(16) float lds[16 * 66];

  const int lane = threadIdx.x;        // 0..31
  const int bcol = lane & 15;          // batch column (N) == A-operand M row
  const int h    = lane >> 4;          // lane half
  const int bg   = blockIdx.x * 16 + bcol;

  // ---- per-row valid length from boolean mask (mask[b,t] = t < len_b) ----
  int len = 0;
  const uint4* mrow = (const uint4*)(mask + (size_t)bg * TLEN);
  #pragma unroll 4
  for (int q = 0; q < TLEN / 16; ++q) {
    uint4 w = mrow[q];
    len += (int)((w.x * 0x01010101u) >> 24);
    len += (int)((w.y * 0x01010101u) >> 24);
    len += (int)((w.z * 0x01010101u) >> 24);
    len += (int)((w.w * 0x01010101u) >> 24);
  }

  // ---- precompute A = E^T tiles (E = exp(trans)), bf16 hi/lo split -------
  // A layout (16-bit 16x32): lane -> M = lane%16, K = 16*(v>>2)+8*(lane>>4)+2*(v&3)+p
  // Only output tiles j in [0,48): cols 0/START are dead (E col == 0) and the
  // STOP column never reaches the output, so tile 3 is dropped entirely.
  // The hi/lo split matters: E quantization error is a *systematic* per-step
  // bias that would accumulate linearly over 1023 steps.
  BF16x16 Ehi[3][2], Elo[3][2];
  #pragma unroll
  for (int tj = 0; tj < 3; ++tj) {
    #pragma unroll
    for (int kc = 0; kc < 2; ++kc) {
      #pragma unroll
      for (int v = 0; v < 8; ++v) {
        unsigned int hp = 0, lp = 0;
        #pragma unroll
        for (int p = 0; p < 2; ++p) {
          int i = 32 * kc + 16 * (v >> 2) + 8 * h + 2 * (v & 3) + p;
          int j = 16 * tj + bcol;
          float e = (i < K0) ? __expf(trans[i * K0 + j]) : 0.f;
          unsigned int hb = f2bf_bits(e);
          unsigned int lb = f2bf_bits(e - bfbits2f(hb));
          hp |= hb << (16 * p);
          lp |= lb << (16 * p);
        }
        Ehi[tj][kc].u[v] = hp;
        Elo[tj][kc].u[v] = lp;
      }
    }
  }

  // ---- zero the dead i in [48,64) rows of the LDS staging tile (once) ----
  {
    int zb = bcol * 66 + 48 + 8 * h;
    #pragma unroll
    for (int r = 0; r < 8; ++r) lds[zb + r] = 0.f;
  }

  // ---- t = 0 init: alpha = exp(feats0 + trans[START,:] - s) --------------
  float alpha[3][8];
  float s;
  {
    const float* f0 = feats + (size_t)bg * TLEN * K0;   // t = 0 row
    const float* tS = trans + START_TAG * K0;
    float x[3][8];
    float m = -3.0e38f;
    #pragma unroll
    for (int tile = 0; tile < 3; ++tile)
      #pragma unroll
      for (int r = 0; r < 8; ++r) {
        int j = 16 * tile + r + 8 * h;                  // j <= 47 < K0
        float xv = f0[j] + tS[j];
        x[tile][r] = xv;
        m = fmaxf(m, xv);
      }
    m = fmaxf(m, __shfl_xor(m, 16, 32));
    s = m;
    #pragma unroll
    for (int tile = 0; tile < 3; ++tile)
      #pragma unroll
      for (int r = 0; r < 8; ++r)
        alpha[tile][r] = __expf(x[tile][r] - m);
  }

  // feats loader in D/C layout: lane(b,h) needs j = 16*tile + r + 8h, r=0..7
  auto load_feats = [&](float (&f)[3][8], int t) {
    const float* frow = feats + ((size_t)bg * TLEN + t) * K0;
    #pragma unroll
    for (int tile = 0; tile < 3; ++tile) {
      int j0 = 16 * tile + 8 * h;                       // j0+7 <= 47, in-bounds
      const float2* p2 = (const float2*)(frow + j0);    // 8B aligned
      #pragma unroll
      for (int q = 0; q < 4; ++q) {
        float2 w = p2[q];
        f[tile][2 * q]     = w.x;
        f[tile][2 * q + 1] = w.y;
      }
    }
  };

  // One scan step. Renorm is an exact invariance (s compensates), and the
  // per-step growth factor is <= ~8e3, so renormalizing every 2nd step keeps
  // everything comfortably inside f32/bf16 range while halving the max-tree,
  // rcp/log and 24-wide rescale on the serial critical path.
  auto step = [&](int t, float (&f)[3][8], const bool do_renorm) {
    // 1) alpha (D layout: row j=16*tile+r+8h, col b=lane%16) -> LDS [b][i]
    {
      int sb = bcol * 66 + 8 * h;
      #pragma unroll
      for (int tile = 0; tile < 3; ++tile)
        #pragma unroll
        for (int r = 0; r < 8; ++r)
          lds[sb + 16 * tile + r] = alpha[tile][r];
    }
    __syncthreads();   // single-wave WG: barrier == S_NOP, acts as fence

    // 2) repack to bf16 B operands (B layout: K = 16*(v>>2)+8h+2*(v&3)+p, N = lane%16)
    BF16x16 B0, B1;
    {
      int lb = bcol * 66 + 8 * h;
      #pragma unroll
      for (int v = 0; v < 8; ++v) {
        float2 a = *(const float2*)&lds[lb + 16 * (v >> 2) + 2 * (v & 3)];        // i in [0,32)
        float2 b = *(const float2*)&lds[lb + 32 + 16 * (v >> 2) + 2 * (v & 3)];   // i in [32,64)
        B0.u[v] = pack_bf16(a.x, a.y);
        B1.u[v] = pack_bf16(b.x, b.y);
      }
    }

    // 3) beta^T = (Ehi + Elo)^T x alpha^T : 12 WMMAs on the critical path
    v8f acc[3];
    #pragma unroll
    for (int tile = 0; tile < 3; ++tile) {
      v8f c = {0.f, 0.f, 0.f, 0.f, 0.f, 0.f, 0.f, 0.f};
      c = __builtin_amdgcn_wmma_f32_16x16x32_bf16(false, Elo[tile][1].v, false, B1.v, (short)0, c, false, false);
      c = __builtin_amdgcn_wmma_f32_16x16x32_bf16(false, Ehi[tile][1].v, false, B1.v, (short)0, c, false, false);
      c = __builtin_amdgcn_wmma_f32_16x16x32_bf16(false, Elo[tile][0].v, false, B0.v, (short)0, c, false, false);
      c = __builtin_amdgcn_wmma_f32_16x16x32_bf16(false, Ehi[tile][0].v, false, B0.v, (short)0, c, false, false);
      acc[tile] = c;
    }

    // 4) multiply by exp(feats), length-mask merge, optional renormalize
    bool act = (t < len);
    if (do_renorm) {
      float m = 0.f;
      #pragma unroll
      for (int tile = 0; tile < 3; ++tile)
        #pragma unroll
        for (int r = 0; r < 8; ++r) {
          float an = acc[tile][r] * __expf(f[tile][r]);
          float av = act ? an : alpha[tile][r];
          alpha[tile][r] = av;
          m = fmaxf(m, av);
        }
      m = fmaxf(m, __shfl_xor(m, 16, 32));   // column spans both lane halves
      float inv = 1.0f / m;
      s += __logf(m);
      #pragma unroll
      for (int tile = 0; tile < 3; ++tile)
        #pragma unroll
        for (int r = 0; r < 8; ++r)
          alpha[tile][r] *= inv;
    } else {
      #pragma unroll
      for (int tile = 0; tile < 3; ++tile)
        #pragma unroll
        for (int r = 0; r < 8; ++r) {
          float an = acc[tile][r] * __expf(f[tile][r]);
          alpha[tile][r] = act ? an : alpha[tile][r];
        }
    }
    __syncthreads();                       // WAR fence before next stores
  };

  // ---- main scan, feats double-buffered 2 steps ahead --------------------
  float fa[3][8], fb[3][8];
  load_feats(fa, 1);
  load_feats(fb, 2);
  int t = 1;
  while (t < TLEN) {
    step(t, fa, /*do_renorm=*/false);
    if (t + 2 < TLEN) load_feats(fa, t + 2);
    ++t;
    if (t >= TLEN) break;
    step(t, fb, /*do_renorm=*/true);
    if (t + 2 < TLEN) load_feats(fb, t + 2);
    ++t;
  }

  // ---- final transition into STOP, deterministic partial per block -------
  // (alpha may be un-renormalized by <= ~8e3 here; (s, alpha) is still an
  //  exact representation, and the log below absorbs it.)
  float dot = 0.f;
  #pragma unroll
  for (int tile = 0; tile < 3; ++tile)
    #pragma unroll
    for (int r = 0; r < 8; ++r) {
      int j = 16 * tile + r + 8 * h;                    // j <= 47
      float es = __expf(trans[j * K0 + STOP_TAG]);
      dot = fmaf(alpha[tile][r], es, dot);
    }
  dot += __shfl_xor(dot, 16, 32);
  float logZ = s + __logf(dot);
  #pragma unroll
  for (int w = 1; w < 16; w <<= 1)
    logZ += __shfl_xor(logZ, w, 16);                    // sum 16 batch rows
  if (lane == 0) partials[blockIdx.x] = logZ;
}

__global__ void crf_reduce_kernel(const float* __restrict__ partials,
                                  float* __restrict__ out)
{
  if (threadIdx.x == 0) {
    float s = 0.f;
    #pragma unroll
    for (int i = 0; i < BATCH / 16; ++i) s += partials[i];
    out[0] = s;
  }
}

extern "C" void kernel_launch(void* const* d_in, const int* in_sizes, int n_in,
                              void* d_out, int out_size, void* d_ws, size_t ws_size,
                              hipStream_t stream) {
  (void)in_sizes; (void)n_in; (void)out_size; (void)ws_size;
  const float*         feats = (const float*)d_in[0];
  const float*         trans = (const float*)d_in[1];
  const unsigned char* mask  = (const unsigned char*)d_in[2];
  float* partials = (float*)d_ws;

  crf_forward_kernel<<<BATCH / 16, 32, 0, stream>>>(feats, trans, mask, partials);
  crf_reduce_kernel<<<1, 32, 0, stream>>>(partials, (float*)d_out);
}
